// MultiHeadAttention_5403068858573
// MI455X (gfx1250) — matrix operand order repaired and checked by reading, not compiled, not run on hardware
//
#include <hip/hip_runtime.h>
#include <stdint.h>

// ---------------- problem constants ----------------
#define BB   4
#define SS   2048
#define DIN  1024
#define HH   16
#define DHD  64

typedef __bf16 bf16_t;
typedef __attribute__((ext_vector_type(16))) __bf16 v16bf;
typedef __attribute__((ext_vector_type(8)))  __bf16 v8bf;
typedef __attribute__((ext_vector_type(8)))  float  v8f;

// ---------------- WMMA helper ----------------
// D(16x16,f32) = A(16x32,bf16) * B(32x16,bf16) + C
__device__ __forceinline__ v8f wmma_bf16(v16bf a, v16bf b, v8f c) {
  return __builtin_amdgcn_wmma_f32_16x16x32_bf16(
      /*neg_a=*/false, a, /*neg_b=*/false, b,
      /*c_mod=*/(short)0, c, /*reuse_a=*/false, /*reuse_b=*/false);
}

// A fragment loader: src is row-major (16 rows, K contiguous, stride elems).
// ISA layout: lanes 0-15 -> M=lane, K {0..7,16..23}; lanes 16-31 -> M=lane-16, K {8..15,24..31}
__device__ __forceinline__ v16bf load_a_frag(const bf16_t* base, int stride, int lane) {
  int m    = lane & 15;
  int half = lane >> 4;
  const bf16_t* p = base + (size_t)m * stride + half * 8;
  union { v16bf v; v8bf h[2]; } u;
  u.h[0] = *(const v8bf*)(p);        // K = half*8 .. +7
  u.h[1] = *(const v8bf*)(p + 16);   // K = half*8+16 .. +23
  return u.v;
}

// B fragment loader: src is column-major-in-K, i.e. mat[col][k] with K contiguous.
// ISA layout: lanes 0-15 -> N=lane, K {0..7,16..23}; lanes 16-31 -> N=lane-16, K {8..15,24..31}
__device__ __forceinline__ v16bf load_b_frag(const bf16_t* base, int col_stride, int lane) {
  int col  = lane & 15;
  int koff = (lane >> 4) * 8;
  const bf16_t* p = base + (size_t)col * col_stride + koff;
  union { v16bf v; v8bf h[2]; } u;
  u.h[0] = *(const v8bf*)(p);
  u.h[1] = *(const v8bf*)(p + 16);
  return u.v;
}

// ---------------- conversion kernels ----------------
__global__ void cvt_f32_bf16_kernel(const float* __restrict__ x,
                                    bf16_t* __restrict__ y, int n) {
  int i = blockIdx.x * blockDim.x + threadIdx.x;
  if (i < n) y[i] = (bf16_t)x[i];
}

// w: [nmat][DIN][64] f32  ->  wt: [nmat][64][DIN] bf16   (K-contiguous per column)
__global__ void cvt_w_transpose_kernel(const float* __restrict__ w,
                                       bf16_t* __restrict__ wt, int total) {
  int i = blockIdx.x * blockDim.x + threadIdx.x;
  if (i >= total) return;
  int e = i % DHD;
  int k = (i / DHD) % DIN;
  int m = i / (DHD * DIN);
  wt[((size_t)m * DHD + e) * DIN + k] = (bf16_t)w[i];
}

// ---------------- QKV projection ----------------
// One wave computes a 32x64 output tile (2 A-frags reuse each B-frag twice).
// k-loop unrolled x2 with ping-pong fragment buffers: loads stay one stage
// ahead of their consuming WMMAs and no register rotation copies are needed.
__global__ __launch_bounds__(32) void qkv_proj_kernel(
    const bf16_t* __restrict__ xbf,   // [B*S, DIN]
    const bf16_t* __restrict__ wt,    // [3][H][64][DIN]
    const float*  __restrict__ biasq, // [H][64]
    const float*  __restrict__ biask,
    const float*  __restrict__ biasv,
    bf16_t* __restrict__ Qbf,         // [B*H][S][64]
    bf16_t* __restrict__ Kbf,         // [B*H][S][64]
    bf16_t* __restrict__ Vt)          // [B*H][64][S]  (transposed)
{
  const int lane = threadIdx.x;
  const int mat  = blockIdx.y;               // 0=Q 1=K 2=V
  const int tile = blockIdx.x;               // over B*H*(S/32)
  const int TS   = SS / 32;
  const int st   = tile % TS;
  const int bh   = tile / TS;
  const int h    = bh % HH;
  const int b    = bh / HH;

  const bf16_t* xrow = xbf + ((size_t)b * SS + (size_t)st * 32) * DIN;
  const bf16_t* w    = wt  + (((size_t)mat * HH + h) * DHD) * DIN;

  v8f acc[8] = {v8f{}, v8f{}, v8f{}, v8f{}, v8f{}, v8f{}, v8f{}, v8f{}};

  // prologue: buffer A holds k = 0
  v16bf a0A = load_a_frag(xrow, DIN, lane);
  v16bf a1A = load_a_frag(xrow + (size_t)16 * DIN, DIN, lane);
  v16bf bA[4];
#pragma unroll
  for (int n = 0; n < 4; ++n)
    bA[n] = load_b_frag(w + (size_t)(n * 16) * DIN, DIN, lane);

  for (int k0 = 0; k0 < DIN; k0 += 64) {
    // start loads for buffer B (k0+32) while buffer A computes
    const int kB = k0 + 32;
    v16bf a0B = load_a_frag(xrow + kB, DIN, lane);
    v16bf a1B = load_a_frag(xrow + (size_t)16 * DIN + kB, DIN, lane);
    v16bf bB[4];
#pragma unroll
    for (int n = 0; n < 4; ++n)
      bB[n] = load_b_frag(w + (size_t)(n * 16) * DIN + kB, DIN, lane);

#pragma unroll
    for (int n = 0; n < 4; ++n) {
      acc[n]     = wmma_bf16(a0A, bA[n], acc[n]);
      acc[4 + n] = wmma_bf16(a1A, bA[n], acc[4 + n]);
    }

    // start loads for buffer A (k0+64; dummy re-load on last iteration)
    const int kA = (k0 + 64 < DIN) ? (k0 + 64) : 0;
    a0A = load_a_frag(xrow + kA, DIN, lane);
    a1A = load_a_frag(xrow + (size_t)16 * DIN + kA, DIN, lane);
#pragma unroll
    for (int n = 0; n < 4; ++n)
      bA[n] = load_b_frag(w + (size_t)(n * 16) * DIN + kA, DIN, lane);

#pragma unroll
    for (int n = 0; n < 4; ++n) {
      acc[n]     = wmma_bf16(a0B, bB[n], acc[n]);
      acc[4 + n] = wmma_bf16(a1B, bB[n], acc[4 + n]);
    }
  }

  const float* bias = (mat == 0 ? biasq : (mat == 1 ? biask : biasv)) + h * DHD;
  const int col   = lane & 15;
  const int rbase = (lane >> 4) * 8;
#pragma unroll
  for (int mb = 0; mb < 2; ++mb) {
#pragma unroll
    for (int n = 0; n < 4; ++n) {
      float bval = bias[n * 16 + col];
#pragma unroll
      for (int i = 0; i < 8; ++i) {
        float val = acc[mb * 4 + n][i] + bval;
        int srow  = st * 32 + mb * 16 + rbase + i;
        int e     = n * 16 + col;
        if (mat == 0)
          Qbf[((size_t)bh * SS + srow) * DHD + e] = (bf16_t)val;
        else if (mat == 1)
          Kbf[((size_t)bh * SS + srow) * DHD + e] = (bf16_t)val;
        else
          Vt[((size_t)bh * DHD + e) * SS + srow] = (bf16_t)val;
      }
    }
  }
}

// ---------------- causal flash attention ----------------
// One wave handles 16 query rows for one (b,h). Key tiles of 32.
__global__ __launch_bounds__(32) void attn_kernel(
    const bf16_t* __restrict__ Qbf,  // [B*H][S][64]
    const bf16_t* __restrict__ Kbf,  // [B*H][S][64]
    const bf16_t* __restrict__ Vt,   // [B*H][64][S]
    bf16_t* __restrict__ Oc)         // [B][S][H*64] concat layout
{
  __shared__ bf16_t pshm[16 * 32];   // P tile staging (C-layout -> A-layout)

  const int lane = threadIdx.x;
  const int tile = blockIdx.x;       // B*H*(S/16)
  const int TS   = SS / 16;
  const int st   = tile % TS;
  const int bh   = tile / TS;
  const int h    = bh % HH;
  const int b    = bh / HH;
  const int qb   = st * 16;

  const int col   = lane & 15;
  const int half  = lane >> 4;
  const int rbase = half * 8;

  const bf16_t* qbase = Qbf + ((size_t)bh * SS + qb) * DHD;
  v16bf aq0 = load_a_frag(qbase + 0,  DHD, lane);   // K 0..31
  v16bf aq1 = load_a_frag(qbase + 32, DHD, lane);   // K 32..63

  v8f o[4] = {v8f{}, v8f{}, v8f{}, v8f{}};
  float m_i[8], l_i[8];
#pragma unroll
  for (int i = 0; i < 8; ++i) { m_i[i] = -3.0e38f; l_i[i] = 0.0f; }

  const float scale  = 0.125f;       // 1/sqrt(64)
  const int   ntiles = (qb + 16 + 31) / 32;

  const bf16_t* kmat = Kbf + (size_t)bh * SS * DHD;
  const bf16_t* vmat = Vt  + (size_t)bh * DHD * SS;

  for (int j = 0; j < ntiles; ++j) {
    const int kb = j * 32;
    const bf16_t* kb0 = kmat + (size_t)kb * DHD;

    // prefetch next key/value tiles toward the WGP while we compute this one
    if (j + 1 < ntiles) {
      const bf16_t* kn = kmat + (size_t)(kb + 32) * DHD;
      const bf16_t* vn = vmat + (kb + 32);
      __builtin_prefetch(kn + (size_t)lane * DHD, 0, 3);          // 32 rows x 128B
      __builtin_prefetch(vn + (size_t)lane * SS, 0, 3);           // V^T rows 0..31
      __builtin_prefetch(vn + (size_t)(lane + 32) * SS, 0, 3);    // V^T rows 32..63
    }

    // scores: 16 queries x 32 keys (two 16-key column halves)
    v8f s0 = v8f{}, s1 = v8f{};
    {
      v16bf b00 = load_b_frag(kb0,      DHD, lane);
      v16bf b01 = load_b_frag(kb0 + 32, DHD, lane);
      const bf16_t* kb1 = kb0 + (size_t)16 * DHD;
      v16bf b10 = load_b_frag(kb1,      DHD, lane);
      v16bf b11 = load_b_frag(kb1 + 32, DHD, lane);
      s0 = wmma_bf16(aq0, b00, s0);
      s0 = wmma_bf16(aq1, b01, s0);
      s1 = wmma_bf16(aq0, b10, s1);
      s1 = wmma_bf16(aq1, b11, s1);
    }

    const bool full = (kb + 31 <= qb);   // tile entirely below diagonal
    float sv0[8], sv1[8];
#pragma unroll
    for (int i = 0; i < 8; ++i) {
      float x0 = s0[i] * scale;
      float x1 = s1[i] * scale;
      if (!full) {
        int row = qb + rbase + i;
        if (kb + col      > row) x0 = -3.0e38f;
        if (kb + 16 + col > row) x1 = -3.0e38f;
      }
      sv0[i] = x0; sv1[i] = x1;
    }

    // online softmax per row (rows live in 16-lane halves)
#pragma unroll
    for (int i = 0; i < 8; ++i) {
      float mx = fmaxf(sv0[i], sv1[i]);
#pragma unroll
      for (int off = 8; off >= 1; off >>= 1)
        mx = fmaxf(mx, __shfl_xor(mx, off, 16));
      float mnew = fmaxf(m_i[i], mx);
      float c    = __expf(m_i[i] - mnew);
      float p0   = __expf(sv0[i] - mnew);
      float p1   = __expf(sv1[i] - mnew);
      float rs   = p0 + p1;
#pragma unroll
      for (int off = 8; off >= 1; off >>= 1)
        rs += __shfl_xor(rs, off, 16);
      l_i[i] = l_i[i] * c + rs;
      m_i[i] = mnew;
#pragma unroll
      for (int n = 0; n < 4; ++n) o[n][i] *= c;
      // stage P (16x32 row-major) in LDS: row = rbase+i, cols col and col+16
      pshm[(rbase + i) * 32 + col]      = (bf16_t)p0;
      pshm[(rbase + i) * 32 + col + 16] = (bf16_t)p1;
    }
    __syncthreads();   // single-wave WG: LDS fence/ordering

    // reload P as A-fragment (16x32)
    v16bf pa;
    {
      union { v16bf v; v8bf hh[2]; } u;
      const bf16_t* p = pshm + (size_t)(lane & 15) * 32 + half * 8;
      u.hh[0] = *(const v8bf*)(p);
      u.hh[1] = *(const v8bf*)(p + 16);
      pa = u.v;
    }
    __syncthreads();

    // O += P * V  (V^T layout -> contiguous B fragments)
    const bf16_t* vb = vmat + kb;
#pragma unroll
    for (int n = 0; n < 4; ++n) {
      v16bf bv = load_b_frag(vb + (size_t)(n * 16) * SS, SS, lane);
      o[n] = wmma_bf16(pa, bv, o[n]);
    }
  }

  // epilogue: normalize and store into concat layout [b][s][h*64+e]
  bf16_t* out = Oc + ((size_t)b * SS + qb) * (HH * DHD) + h * DHD;
#pragma unroll
  for (int i = 0; i < 8; ++i) {
    float inv = 1.0f / l_i[i];
    int row = rbase + i;
#pragma unroll
    for (int n = 0; n < 4; ++n)
      out[(size_t)row * (HH * DHD) + n * 16 + col] = (bf16_t)(o[n][i] * inv);
  }
}

// ---------------- output projection ----------------
// out(32x64 f32) = Oc(32x1024 bf16) * Wo^T-layout(64x1024 bf16) + bo
// Same 32-row tile + ping-pong pipelining as the QKV projection.
__global__ __launch_bounds__(32) void out_proj_kernel(
    const bf16_t* __restrict__ Oc,   // [B*S][1024]
    const bf16_t* __restrict__ Wot,  // [64][1024]
    const float*  __restrict__ bo,   // [64]
    float* __restrict__ out)         // [B*S][64]
{
  const int lane = threadIdx.x;
  const int tile = blockIdx.x;       // B*S/32
  const int KD   = HH * DHD;         // 1024
  const bf16_t* arow = Oc + (size_t)tile * 32 * KD;

  v8f acc[8] = {v8f{}, v8f{}, v8f{}, v8f{}, v8f{}, v8f{}, v8f{}, v8f{}};

  v16bf a0A = load_a_frag(arow, KD, lane);
  v16bf a1A = load_a_frag(arow + (size_t)16 * KD, KD, lane);
  v16bf bA[4];
#pragma unroll
  for (int n = 0; n < 4; ++n)
    bA[n] = load_b_frag(Wot + (size_t)(n * 16) * KD, KD, lane);

  for (int k0 = 0; k0 < KD; k0 += 64) {
    const int kB = k0 + 32;
    v16bf a0B = load_a_frag(arow + kB, KD, lane);
    v16bf a1B = load_a_frag(arow + (size_t)16 * KD + kB, KD, lane);
    v16bf bB[4];
#pragma unroll
    for (int n = 0; n < 4; ++n)
      bB[n] = load_b_frag(Wot + (size_t)(n * 16) * KD + kB, KD, lane);

#pragma unroll
    for (int n = 0; n < 4; ++n) {
      acc[n]     = wmma_bf16(a0A, bA[n], acc[n]);
      acc[4 + n] = wmma_bf16(a1A, bA[n], acc[4 + n]);
    }

    const int kA = (k0 + 64 < KD) ? (k0 + 64) : 0;
    a0A = load_a_frag(arow + kA, KD, lane);
    a1A = load_a_frag(arow + (size_t)16 * KD + kA, KD, lane);
#pragma unroll
    for (int n = 0; n < 4; ++n)
      bA[n] = load_b_frag(Wot + (size_t)(n * 16) * KD + kA, KD, lane);

#pragma unroll
    for (int n = 0; n < 4; ++n) {
      acc[n]     = wmma_bf16(a0B, bB[n], acc[n]);
      acc[4 + n] = wmma_bf16(a1B, bB[n], acc[4 + n]);
    }
  }

  const int col   = lane & 15;
  const int rbase = (lane >> 4) * 8;
#pragma unroll
  for (int mb = 0; mb < 2; ++mb) {
#pragma unroll
    for (int n = 0; n < 4; ++n) {
      float bval = bo[n * 16 + col];
#pragma unroll
      for (int i = 0; i < 8; ++i) {
        int row = tile * 32 + mb * 16 + rbase + i;
        out[(size_t)row * DHD + n * 16 + col] = acc[mb * 4 + n][i] + bval;
      }
    }
  }
}

// ---------------- host launch ----------------
static inline size_t align256(size_t x) { return (x + 255) & ~(size_t)255; }

extern "C" void kernel_launch(void* const* d_in, const int* in_sizes, int n_in,
                              void* d_out, int out_size, void* d_ws, size_t ws_size,
                              hipStream_t stream) {
  (void)in_sizes; (void)n_in; (void)out_size; (void)ws_size;

  const float* x  = (const float*)d_in[0];   // [B,S,DIN]
  const float* Wq = (const float*)d_in[1];   // [H,DIN,64]
  const float* bq = (const float*)d_in[2];   // [H,64]
  const float* Wk = (const float*)d_in[3];
  const float* bk = (const float*)d_in[4];
  const float* Wv = (const float*)d_in[5];
  const float* bv = (const float*)d_in[6];
  const float* Wo = (const float*)d_in[7];   // [1024,64]
  const float* bo = (const float*)d_in[8];   // [64]
  float* out = (float*)d_out;                // [B,S,64]

  const size_t nX  = (size_t)BB * SS * DIN;       // 8388608
  const size_t nW  = (size_t)HH * DHD * DIN;      // 1048576 (per matrix)
  const size_t nWo = (size_t)DHD * DIN;           // 65536
  const size_t nQ  = (size_t)BB * HH * SS * DHD;  // 8388608
  const size_t nOc = (size_t)BB * SS * HH * DHD;  // 8388608

  uint8_t* ws = (uint8_t*)d_ws;
  size_t off = 0;
  bf16_t* xbf = (bf16_t*)(ws + off); off = align256(off + nX  * 2);
  bf16_t* wt  = (bf16_t*)(ws + off); off = align256(off + 3 * nW * 2);
  bf16_t* wot = (bf16_t*)(ws + off); off = align256(off + nWo * 2);
  bf16_t* Qbf = (bf16_t*)(ws + off); off = align256(off + nQ  * 2);
  bf16_t* Kbf = (bf16_t*)(ws + off); off = align256(off + nQ  * 2);
  bf16_t* Vt  = (bf16_t*)(ws + off); off = align256(off + nQ  * 2);
  bf16_t* Oc  = (bf16_t*)(ws + off); off = align256(off + nOc * 2);
  // total workspace use ~86 MB

  // 1) convert x to bf16 (x stays L2-resident afterwards: 16 MB vs 192 MB L2)
  cvt_f32_bf16_kernel<<<(int)((nX + 255) / 256), 256, 0, stream>>>(x, xbf, (int)nX);
  // 2) transpose+convert weights into K-contiguous B-fragment layouts
  cvt_w_transpose_kernel<<<(int)((nW + 255) / 256), 256, 0, stream>>>(Wq, wt + 0 * nW, (int)nW);
  cvt_w_transpose_kernel<<<(int)((nW + 255) / 256), 256, 0, stream>>>(Wk, wt + 1 * nW, (int)nW);
  cvt_w_transpose_kernel<<<(int)((nW + 255) / 256), 256, 0, stream>>>(Wv, wt + 2 * nW, (int)nW);
  cvt_w_transpose_kernel<<<(int)((nWo + 255) / 256), 256, 0, stream>>>(Wo, wot, (int)nWo);
  // 3) QKV projections (WMMA, 32-row tiles, ping-pong pipelined)
  {
    dim3 grid(BB * HH * (SS / 32), 3, 1);
    qkv_proj_kernel<<<grid, 32, 0, stream>>>(xbf, wt, bq, bk, bv, Qbf, Kbf, Vt);
  }
  // 4) causal flash attention (WMMA + global_prefetch of next tiles)
  attn_kernel<<<BB * HH * (SS / 16), 32, 0, stream>>>(Qbf, Kbf, Vt, Oc);
  // 5) output projection (WMMA, 32-row tiles, ping-pong pipelined)
  out_proj_kernel<<<BB * (SS / 32), 32, 0, stream>>>(Oc, wot, bo, out);
}
